// MultiHeadAttention_71829033058396
// MI455X (gfx1250) — matrix operand order repaired and checked by reading, not compiled, not run on hardware
//
#include <hip/hip_runtime.h>
#include <hip/hip_bf16.h>
#include <cstdint>

// Problem constants (match reference)
#define B_  4
#define N_  2048
#define M_  2048
#define C_  512
#define H_  8
#define DH_ 64
#define MP_ 2056   // padded LDS row stride (floats), keeps 16B alignment

typedef __attribute__((ext_vector_type(16))) __bf16 v16bf;
typedef __attribute__((ext_vector_type(8)))  __bf16 bf16x8;
typedef __attribute__((ext_vector_type(4)))  __bf16 bf16x4;
typedef __attribute__((ext_vector_type(8)))  float  v8f;

// Hardware f32->bf16 conversion (v_cvt_*_bf16 on gfx1250)
static __device__ __forceinline__ __bf16 f2bf(float f) { return (__bf16)f; }

// Combine two contiguous 8-element halves into a 16-element fragment register
static __device__ __forceinline__ v16bf cat8(bf16x8 lo, bf16x8 hi) {
  return __builtin_shufflevector(lo, hi, 0, 1, 2, 3, 4, 5, 6, 7,
                                 8, 9, 10, 11, 12, 13, 14, 15);
}

// ---------------------------------------------------------------------------
// Elementwise f32 -> bf16 (pre-pass so GEMMs read clean bf16, no cvt in loop)
// ---------------------------------------------------------------------------
__global__ void cvt_f32_bf16_kernel(const float* __restrict__ x,
                                    __bf16* __restrict__ y, size_t n) {
  size_t i = ((size_t)blockIdx.x * blockDim.x + threadIdx.x) * 4;
  const size_t stride = (size_t)gridDim.x * blockDim.x * 4;
  for (; i < n; i += stride) {
    const float4 v = *(const float4*)(x + i);
    bf16x4 o;
    o[0] = f2bf(v.x); o[1] = f2bf(v.y); o[2] = f2bf(v.z); o[3] = f2bf(v.w);
    *(bf16x4*)(y + i) = o;
  }
}

// ---------------------------------------------------------------------------
// W (f32, [k][c]) -> Wt (bf16, [c][k]) via LDS tile transpose.
// Block = 256 threads, 16x16 tile.
// ---------------------------------------------------------------------------
__global__ void wtrans_bf16_kernel(const float* __restrict__ W,
                                   __bf16* __restrict__ Wt) {
  __shared__ float tile[16][17];
  const int k0 = blockIdx.x * 16;
  const int c0 = blockIdx.y * 16;
  const int tx = threadIdx.x & 15;
  const int ty = threadIdx.x >> 4;
  tile[ty][tx] = W[(size_t)(k0 + ty) * C_ + c0 + tx];
  __syncthreads();
  Wt[(size_t)(c0 + ty) * C_ + k0 + tx] = f2bf(tile[tx][ty]);
}

// ---------------------------------------------------------------------------
// Projection GEMM: Y = Xbf @ Wt^T + bias (Xbf bf16 [b][s][C], Wt bf16 [c][k]).
//   transposed==0 : Y[b][h][s][dh]   (Q, K layout)
//   transposed==1 : Y[b][h][dh][s]   (V transposed)
// Block = 128 threads (4 waves). Block tile = 16 rows x 512 cols.
// ---------------------------------------------------------------------------
__global__ void proj_bf16_wmma_kernel(const __bf16* __restrict__ Xbf,
                                      const float* __restrict__ bias,
                                      const __bf16* __restrict__ Wt,
                                      __bf16* __restrict__ Y,
                                      int S, int transposed) {
  const int tid  = threadIdx.x;
  const int lane = tid & 31;
  const int wave = tid >> 5;
  const int l15  = lane & 15;
  const bool hih = lane >= 16;
  const int s0   = blockIdx.x * 16;
  const int bb   = blockIdx.y;

  const __bf16* Xrow = Xbf + ((size_t)bb * S + s0 + l15) * C_;

  v8f acc[8] = {};

  for (int kc = 0; kc < C_; kc += 32) {
    // A fragment: lane<16 holds K {0..15}, lane>=16 holds {16..31}
    const int kb = kc + (hih ? 16 : 0);
    const v16bf a = cat8(*(const bf16x8*)(Xrow + kb),
                         *(const bf16x8*)(Xrow + kb + 8));
    const int krb = kc + (hih ? 16 : 0);
#pragma unroll
    for (int t = 0; t < 8; ++t) {
      const int col = (wave * 8 + t) * 16 + l15;
      // B fragment: lane = column, 16 consecutive K -> contiguous in Wt
      const v16bf bf = *(const v16bf*)(Wt + (size_t)col * C_ + krb);
      acc[t] = __builtin_amdgcn_wmma_f32_16x16x32_bf16(
          false, a, false, bf, (short)0, acc[t], false, false);
    }
  }

  // Bias add + store (C/D layout: row = r + 8*(lane>=16), col = lane&15)
#pragma unroll
  for (int t = 0; t < 8; ++t) {
    const int col = (wave * 8 + t) * 16 + l15;
    const int hh  = col >> 6;
    const int dh  = col & 63;
    const float bv = bias[col];
#pragma unroll
    for (int r = 0; r < 8; ++r) {
      const int srow = s0 + r + (hih ? 8 : 0);
      const float val = acc[t][r] + bv;
      if (!transposed)
        Y[((size_t)(bb * H_ + hh) * S + srow) * DH_ + dh] = f2bf(val);
      else
        Y[((size_t)(bb * H_ + hh) * DH_ + dh) * (size_t)S + srow] = f2bf(val);
    }
  }
}

// ---------------------------------------------------------------------------
// Fused attention: per workgroup = one (b, h, 16-row n-tile).
// 4 waves, 128 threads. Full 16 x 2048 score strip in LDS (CDNA5: 320KB/WGP).
//   Phase 1: S = (Q Kᵀ)/8 * factors * key_w, masks -> -inf   (WMMA bf16)
//   Phase 2: block softmax in LDS; write normalized attn to HBM once
//   Phase 3: hidden = (exp-strip · V) / rowsum               (WMMA bf16)
// ---------------------------------------------------------------------------
__global__ void attn_fused_wmma_kernel(const __bf16* __restrict__ Qh,
                                       const __bf16* __restrict__ Kh,
                                       const __bf16* __restrict__ Vt,
                                       const float* __restrict__ factors,
                                       const float* __restrict__ keyw,
                                       const unsigned char* __restrict__ kmask,
                                       const unsigned char* __restrict__ amask,
                                       float* __restrict__ hidden,
                                       float* __restrict__ attn) {
  extern __shared__ float smem[];
  float* sc   = smem;                 // 16 * MP_ : score / exp strip
  float* red  = sc + 16 * MP_;        // 16 * 8   : reduction scratch
  float* rmax = red + 128;            // 16
  float* rsum = rmax + 16;            // 16
  float* hidb = rsum + 16;            // 4 * 16 * 64 : per-wave hidden partials

  const int tid  = threadIdx.x;
  const int lane = tid & 31;
  const int wave = tid >> 5;
  const int l15  = lane & 15;
  const bool hih = lane >= 16;
  const int n0 = blockIdx.x * 16;
  const int h  = blockIdx.y;
  const int b  = blockIdx.z;
  const size_t bh = (size_t)b * H_ + h;

  // ---- Q A-fragments (held in VGPRs): DH=64 -> 2 frags
  const __bf16* qrow = Qh + (bh * N_ + n0 + l15) * DH_;
  const int kba = hih ? 16 : 0;
  const v16bf aq0 = cat8(*(const bf16x8*)(qrow + kba),
                         *(const bf16x8*)(qrow + kba + 8));
  const v16bf aq1 = cat8(*(const bf16x8*)(qrow + 32 + kba),
                         *(const bf16x8*)(qrow + 32 + kba + 8));

  // ---- Phase 1: scores into LDS, 16-col tiles round-robin across waves
  const int kbb = hih ? 16 : 0;
  for (int mt = wave; mt < M_ / 16; mt += 4) {
    const int m0 = mt * 16;
    const __bf16* krow = Kh + (bh * M_ + m0 + l15) * DH_;
    if (mt + 4 < M_ / 16)
      __builtin_prefetch(Kh + (bh * M_ + (mt + 4) * 16 + l15) * DH_, 0, 1);
    const v16bf b0 = *(const v16bf*)(krow + kbb);
    const v16bf b1 = *(const v16bf*)(krow + 32 + kbb);
    v8f c = {};
    c = __builtin_amdgcn_wmma_f32_16x16x32_bf16(false, aq0, false, b0, (short)0, c, false, false);
    c = __builtin_amdgcn_wmma_f32_16x16x32_bf16(false, aq1, false, b1, (short)0, c, false, false);

    const int col = m0 + l15;
    const float kwv = keyw[(size_t)b * M_ + col];
    const bool  km  = kmask[(size_t)b * M_ + col] != 0;
#pragma unroll
    for (int r = 0; r < 8; ++r) {
      const int row = r + (hih ? 8 : 0);
      const size_t nm = ((size_t)b * N_ + n0 + row) * M_ + col;
      float s = c[r] * 0.125f * factors[nm] * kwv;
      if (km || amask[nm]) s = -__builtin_inff();
      sc[row * MP_ + col] = s;
    }
  }
  __syncthreads();

  // ---- Phase 2: softmax over M (16 rows x 2048), 8 threads per row
  const int rr  = tid >> 3;
  const int sub = tid & 7;
  float mx = -__builtin_inff();
  for (int m = sub; m < M_; m += 8) mx = fmaxf(mx, sc[rr * MP_ + m]);
  red[rr * 8 + sub] = mx;
  __syncthreads();
  if (sub == 0) {
    float v = red[rr * 8];
#pragma unroll
    for (int i = 1; i < 8; ++i) v = fmaxf(v, red[rr * 8 + i]);
    rmax[rr] = v;
  }
  __syncthreads();
  const float rm = rmax[rr];
  float psum = 0.f;
  for (int m = sub; m < M_; m += 8) {
    float e = __expf(sc[rr * MP_ + m] - rm);
    sc[rr * MP_ + m] = e;   // keep UNNORMALIZED exp in LDS for P·V
    psum += e;
  }
  red[rr * 8 + sub] = psum;
  __syncthreads();
  if (sub == 0) {
    float v = 0.f;
#pragma unroll
    for (int i = 0; i < 8; ++i) v += red[rr * 8 + i];
    rsum[rr] = v;
  }
  __syncthreads();

  // Write normalized attn to HBM (coalesced: 128 threads sweep contiguous m)
  for (int row = 0; row < 16; ++row) {
    const float inv = 1.f / rsum[row];
    float* dst = attn + (bh * N_ + n0 + row) * M_;
    for (int m = tid; m < M_; m += 128) dst[m] = sc[row * MP_ + m] * inv;
  }

  // ---- Phase 3: hidden = exp-strip · V (normalize at the end).
  v8f acc[4] = {};
  const int plo = hih ? 16 : 0;
  const int phi = hih ? 24 : 8;
  for (int mc = wave; mc < M_ / 32; mc += 4) {
    const int m0 = mc * 32;
    const float* erow = sc + l15 * MP_ + m0;
    v16bf ap;
#pragma unroll
    for (int i = 0; i < 8; ++i) {
      ap[i]     = f2bf(erow[plo + i]);
      ap[8 + i] = f2bf(erow[phi + i]);
    }
#pragma unroll
    for (int d = 0; d < 4; ++d) {
      const v16bf bv = *(const v16bf*)(Vt + (bh * DH_ + d * 16 + l15) * M_ + m0 + kbb);
      acc[d] = __builtin_amdgcn_wmma_f32_16x16x32_bf16(
          false, ap, false, bv, (short)0, acc[d], false, false);
    }
  }

  // Cross-wave reduction of 16x64 hidden partials via LDS
#pragma unroll
  for (int d = 0; d < 4; ++d)
#pragma unroll
    for (int r = 0; r < 8; ++r)
      hidb[(wave * 16 + r + (hih ? 8 : 0)) * 64 + d * 16 + l15] = acc[d][r];
  __syncthreads();

  for (int idx = tid; idx < 16 * 64; idx += 128) {
    const int row = idx >> 6;
    const int dh  = idx & 63;
    float v = 0.f;
#pragma unroll
    for (int w = 0; w < 4; ++w) v += hidb[(w * 16 + row) * 64 + dh];
    v *= 1.f / rsum[row];
    hidden[((size_t)b * N_ + n0 + row) * C_ + h * DH_ + dh] = v;
  }
}

// ---------------------------------------------------------------------------
extern "C" void kernel_launch(void* const* d_in, const int* in_sizes, int n_in,
                              void* d_out, int out_size, void* d_ws, size_t ws_size,
                              hipStream_t stream) {
  (void)in_sizes; (void)n_in; (void)out_size; (void)ws_size;

  const float* input_q = (const float*)d_in[0];
  const float* input_k = (const float*)d_in[1];
  const float* input_v = (const float*)d_in[2];
  const float* key_w   = (const float*)d_in[3];
  const unsigned char* key_m = (const unsigned char*)d_in[4];   // bool (1B)
  const float* att_f   = (const float*)d_in[5];
  const unsigned char* att_m = (const unsigned char*)d_in[6];   // bool (1B)
  const float* Wq = (const float*)d_in[7];
  const float* bq = (const float*)d_in[8];
  const float* Wk = (const float*)d_in[9];
  const float* bk = (const float*)d_in[10];
  const float* Wv = (const float*)d_in[11];
  const float* bv = (const float*)d_in[12];

  float* hidden = (float*)d_out;                       // (B,N,C)
  float* attn   = hidden + (size_t)B_ * N_ * C_;       // (B,H,N,M)

  // Workspace layout (bf16 elements):
  //   Qh (B,H,N,DH) | Kh (B,H,M,DH) | Vt (B,H,DH,M)   -- 4M elems each
  //   Xq | Xk | Xv (B,S,C bf16)                        -- 4M elems each
  //   Wtq | Wtk | Wtv (C,C bf16)                       -- 256K elems each
  const size_t TENS = (size_t)B_ * H_ * N_ * DH_;      // == B*N*C
  __bf16* Qh  = (__bf16*)d_ws;
  __bf16* Kh  = Qh + TENS;
  __bf16* Vt  = Kh + TENS;
  __bf16* Xq  = Vt + TENS;
  __bf16* Xk  = Xq + TENS;
  __bf16* Xv  = Xk + TENS;
  __bf16* Wtq = Xv + TENS;
  __bf16* Wtk = Wtq + (size_t)C_ * C_;
  __bf16* Wtv = Wtk + (size_t)C_ * C_;

  // --- pre-pass: convert activations, transpose+convert weights
  const size_t nX = (size_t)B_ * N_ * C_;
  cvt_f32_bf16_kernel<<<4096, 256, 0, stream>>>(input_q, Xq, nX);
  cvt_f32_bf16_kernel<<<4096, 256, 0, stream>>>(input_k, Xk, nX);
  cvt_f32_bf16_kernel<<<4096, 256, 0, stream>>>(input_v, Xv, nX);
  dim3 wt_grid(C_ / 16, C_ / 16);
  wtrans_bf16_kernel<<<wt_grid, 256, 0, stream>>>(Wq, Wtq);
  wtrans_bf16_kernel<<<wt_grid, 256, 0, stream>>>(Wk, Wtk);
  wtrans_bf16_kernel<<<wt_grid, 256, 0, stream>>>(Wv, Wtv);

  // --- projections (WMMA)
  dim3 blk(128);
  proj_bf16_wmma_kernel<<<dim3(N_ / 16, B_), blk, 0, stream>>>(Xq, bq, Wtq, Qh, N_, 0);
  proj_bf16_wmma_kernel<<<dim3(M_ / 16, B_), blk, 0, stream>>>(Xk, bk, Wtk, Kh, M_, 0);
  proj_bf16_wmma_kernel<<<dim3(M_ / 16, B_), blk, 0, stream>>>(Xv, bv, Wtv, Vt, M_, 1);

  // --- fused attention (WMMA + LDS softmax)
  const size_t smem =
      (size_t)(16 * MP_ + 128 + 16 + 16 + 4 * 16 * 64) * sizeof(float); // ~145 KB
  hipFuncSetAttribute((const void*)attn_fused_wmma_kernel,
                      hipFuncAttributeMaxDynamicSharedMemorySize, (int)smem);
  attn_fused_wmma_kernel<<<dim3(N_ / 16, H_, B_), blk, smem, stream>>>(
      Qh, Kh, Vt, att_f, key_w, key_m, att_m, hidden, attn);
}
